// PerNodeGGNN_65755949301928
// MI455X (gfx1250) — matrix operand order repaired and checked by reading, not compiled, not run on hardware
//
#include <hip/hip_runtime.h>
#include <hip/hip_bf16.h>
#include <stdint.h>

typedef __bf16 bf16_t;
typedef __attribute__((ext_vector_type(16))) __bf16 v16bf;
typedef __attribute__((ext_vector_type(8)))  float  v8f;
typedef __attribute__((ext_vector_type(4)))  unsigned int v4u;
typedef __attribute__((ext_vector_type(8)))  int v8i;
typedef __attribute__((ext_vector_type(4)))  int v4i;

#ifndef __has_builtin
#define __has_builtin(x) 0
#endif
#if __has_builtin(__builtin_amdgcn_tensor_load_to_lds) && __has_builtin(__builtin_amdgcn_s_wait_tensorcnt)
#define HAVE_TDM 1
#else
#define HAVE_TDM 0
#endif

struct Q32 { uint4 a, b; };

static __device__ inline v16bf make_frag(uint4 a, uint4 b) {
    Q32 q; q.a = a; q.b = b;
    return __builtin_bit_cast(v16bf, q);
}

// A fragment (16x32 bf16, M x K), rows row0..row0+15 of a row-major [*, ld] matrix.
// ISA layout: lanes 0-15 hold K = kb+[0..7] and kb+[16..23]; lanes 16-31 hold +8.
static __device__ inline v16bf load_frag_a(const bf16_t* base, int row0, int ld, int kb, int lane) {
    int g = lane >> 4;
    int r = lane & 15;
    const bf16_t* p = base + (size_t)(row0 + r) * ld + kb;
    uint4 c0 = *(const uint4*)(p + 8 * g);
    uint4 c1 = *(const uint4*)(p + 16 + 8 * g);
    return make_frag(c0, c1);
}

// B fragment (32x16 bf16, K x N) sourced from B^T stored row-major [out, in]
// (row = output column, K contiguous). Lanes 0-15: K = kb+[0..7] and kb+[16..23]; lanes 16-31 hold +8.
static __device__ inline v16bf load_frag_b(const bf16_t* base, int col0, int ld, int kb, int lane) {
    int g = lane >> 4;
    int n = lane & 15;
    const bf16_t* p = base + (size_t)(col0 + n) * ld + kb;
    uint4 c0 = *(const uint4*)(p + 8 * g);
    uint4 c1 = *(const uint4*)(p + 16 + 8 * g);
    return make_frag(c0, c1);
}

static __device__ inline v8f wmma_bf16(v16bf a, v16bf b, v8f c) {
    return __builtin_amdgcn_wmma_f32_16x16x32_bf16(false, a, false, b, (short)0, c, false, false);
}

static __device__ inline unsigned bfbits(float f) {
    bf16_t h = (bf16_t)f;
    return (unsigned)__builtin_bit_cast(unsigned short, h);
}

static __device__ inline float sigmoidf_(float x) { return 1.0f / (1.0f + __expf(-x)); }

#define SS 256   // state size
#define ANN_ 128
#define CATW 384
#define MLPH 256
#define OUTW 128

#if HAVE_TDM
// One-shot 2D TDM: rows_avail rows of row_elems bf16 elements (contiguous rows),
// tile tile_rows x row_elems, OOB rows zero-filled by hardware.
static __device__ inline void tdm_load_2d_bf16(const void* gptr, unsigned lds_off,
                                               unsigned rows_avail, unsigned tile_rows,
                                               unsigned row_elems) {
    unsigned long long ga = (unsigned long long)(uintptr_t)gptr;
    v4u g0 = { 1u,                                   // count=1, is_restore=0, gather=0
               lds_off,                              // lds_addr (bytes)
               (unsigned)ga,                         // global_addr[31:0]
               (unsigned)((ga >> 32) & 0x01FFFFFFu) | 0x80000000u }; // addr[56:32] | type=2
    v8i g1;
    g1[0] = (int)(1u << 16);                                              // data_size=2B
    g1[1] = (int)((row_elems & 0xFFFFu) << 16);                           // tensor_dim0 lo
    g1[2] = (int)((row_elems >> 16) | ((rows_avail & 0xFFFFu) << 16));    // dim0 hi | dim1 lo
    g1[3] = (int)((rows_avail >> 16) | ((row_elems & 0xFFFFu) << 16));    // dim1 hi | tile_dim0
    g1[4] = (int)(tile_rows & 0xFFFFu);                                   // tile_dim1 (tile_dim2=0)
    g1[5] = (int)row_elems;                                               // tensor_dim0_stride lo
    g1[6] = 0;
    g1[7] = 0;
    v4i z4 = { 0, 0, 0, 0 };
    v8i z8 = { 0, 0, 0, 0, 0, 0, 0, 0 };
    __builtin_amdgcn_tensor_load_to_lds(g0, g1, z4, z4, z8, 0);
}
static __device__ inline void tdm_wait() { __builtin_amdgcn_s_wait_tensorcnt(0); }
#endif

// ---------------- weight conversion: f32 -> bf16 (W transposed per layer) ----------
__global__ void convert_weights_kernel(const float* W, const float* wih, const float* whh,
                                       const float* l0w, const float* l1w,
                                       bf16_t* WT, bf16_t* wihb, bf16_t* whhb,
                                       bf16_t* l0wb, bf16_t* l1wb) {
    int id = blockIdx.x * 256 + threadIdx.x;
    if (id < 5 * SS * SS) {                    // W: [5,S,S] -> WT: [5, j, k] = W[l,k,j]
        int l = id >> 16, rem = id & 65535;
        int j = rem >> 8, k = rem & 255;
        WT[id] = (bf16_t)W[(l << 16) + (k << 8) + j];
        return;
    }
    int i = id - 5 * SS * SS;
    if (i < 3 * SS * SS) { wihb[i] = (bf16_t)wih[i]; return; }
    i -= 3 * SS * SS;
    if (i < 3 * SS * SS) { whhb[i] = (bf16_t)whh[i]; return; }
    i -= 3 * SS * SS;
    if (i < MLPH * CATW) { l0wb[i] = (bf16_t)l0w[i]; return; }
    i -= MLPH * CATW;
    if (i < OUTW * MLPH) { l1wb[i] = (bf16_t)l1w[i]; return; }
}

// ---------------- h init: zero-pad annotations, also make x_bf16 ----------------
__global__ void init_h_kernel(const float* x, float* hf, bf16_t* hb, bf16_t* xb, int N) {
    int id = blockIdx.x * 256 + threadIdx.x;
    int n = id >> 8, c = id & 255;
    if (n >= N) return;
    float v = (c < ANN_) ? x[(size_t)n * ANN_ + c] : 0.0f;
    hf[id] = v;
    hb[id] = (bf16_t)v;
    if (c < ANN_) xb[(size_t)n * ANN_ + c] = (bf16_t)v;
}

// ---------------- m = h @ W_l  (bf16 in, bf16 out), A and B both TDM-staged in LDS ----
__global__ void __launch_bounds__(128) gemm_m_kernel(const bf16_t* __restrict__ hb,
                                                     const bf16_t* __restrict__ WlT,
                                                     bf16_t* __restrict__ mb, int N) {
    __shared__ bf16_t Atile[64 * SS];
    __shared__ bf16_t Btile[SS * SS];
    int t = threadIdx.x;
    int n0 = blockIdx.x * 64;
#if HAVE_TDM
    if (t < 32) {
        tdm_load_2d_bf16(hb + (size_t)n0 * SS, (unsigned)(uintptr_t)Atile,
                         (unsigned)(N - n0), 64u, (unsigned)SS);
        tdm_load_2d_bf16(WlT, (unsigned)(uintptr_t)Btile, (unsigned)SS, (unsigned)SS,
                         (unsigned)SS);
        tdm_wait();
    }
    __syncthreads();
#else
#pragma unroll
    for (int i = 0; i < 16; ++i) {
        int q = t + i * 128;
        int row = q >> 5, c = q & 31;
        int gr = n0 + row; if (gr >= N) gr = N - 1;
        ((uint4*)Atile)[q] = ((const uint4*)(hb + (size_t)gr * SS))[c];
    }
#pragma unroll
    for (int i = 0; i < 64; ++i) {
        int q = t + i * 128;
        ((uint4*)Btile)[q] = ((const uint4*)WlT)[q];
    }
    __syncthreads();
#endif
    int wave = t >> 5, lane = t & 31;
    int rowbase = wave * 16;
    int rh = lane >> 4, col16 = lane & 15;
    bool full = (n0 + 64 <= N);                 // uniform: all 64 rows valid
    for (int ct = 0; ct < 16; ++ct) {
        v8f acc = {};
        v16bf b = load_frag_b(Btile, ct * 16, SS, 0, lane);
#pragma unroll
        for (int kk = 0; kk < 8; ++kk) {
            v16bf bn = (kk < 7) ? load_frag_b(Btile, ct * 16, SS, (kk + 1) * 32, lane) : b;
            v16bf a = load_frag_a(Atile, rowbase, SS, kk * 32, lane);
            acc = wmma_bf16(a, b, acc);
            b = bn;
        }
        int col = ct * 16 + col16;
        bf16_t* mp = mb + (size_t)(n0 + rowbase + 8 * rh) * SS + col;
        if (full) {
#pragma unroll
            for (int i = 0; i < 8; ++i) mp[(size_t)i * SS] = (bf16_t)acc[i];
        } else {
#pragma unroll
            for (int i = 0; i < 8; ++i) {
                int row = n0 + rowbase + i + 8 * rh;
                if (row < N) mb[(size_t)row * SS + col] = (bf16_t)acc[i];
            }
        }
    }
}

// ---------------- edge scatter: agg[dst] += m[src]  (f32 atomics, L2-resident) ------
__global__ void __launch_bounds__(256) scatter_kernel(const int* __restrict__ ei,
                                                      const bf16_t* __restrict__ mb,
                                                      float* __restrict__ agg, int E) {
    int gid = blockIdx.x * 256 + threadIdx.x;
    int e = gid >> 5;
    if (e >= E) return;
    int lane = gid & 31;
    int src = ei[e];
    int dst = ei[E + e];
    uint4 q = *(const uint4*)(mb + (size_t)src * SS + lane * 8);
    const bf16_t* h8 = (const bf16_t*)&q;
    float* ap = agg + (size_t)dst * SS + lane * 8;
#pragma unroll
    for (int j = 0; j < 8; ++j) unsafeAtomicAdd(ap + j, (float)h8[j]);
}

// ---------------- fused GRU: h_new = GRU(agg, h) ----------------
__global__ void __launch_bounds__(128) gru_kernel(const float* __restrict__ agg,
                                                  const bf16_t* __restrict__ hb,
                                                  const float* __restrict__ hf,
                                                  const bf16_t* __restrict__ wihb,
                                                  const bf16_t* __restrict__ whhb,
                                                  const float* __restrict__ bih,
                                                  const float* __restrict__ bhh,
                                                  float* __restrict__ hf_out,
                                                  bf16_t* __restrict__ hb_out, int N) {
    __shared__ bf16_t Ags[64 * SS];
    __shared__ bf16_t Hs[64 * SS];
    int t = threadIdx.x;
    int n0 = blockIdx.x * 64;
#if HAVE_TDM
    if (t < 32)  // DMA the h tile while all threads convert the agg tile below
        tdm_load_2d_bf16(hb + (size_t)n0 * SS, (unsigned)(uintptr_t)Hs,
                         (unsigned)(N - n0), 64u, (unsigned)SS);
#endif
    // agg tile: f32 -> bf16 into LDS (4096 float4, 32 per thread; packed uint2 stores)
#pragma unroll
    for (int i = 0; i < 32; ++i) {
        int q = t + i * 128;
        int row = q >> 6, c4 = q & 63;
        int gr = n0 + row; if (gr >= N) gr = N - 1;
        float4 v = ((const float4*)(agg + (size_t)gr * SS))[c4];
        uint2 u;
        u.x = bfbits(v.x) | (bfbits(v.y) << 16);
        u.y = bfbits(v.z) | (bfbits(v.w) << 16);
        ((uint2*)Ags)[q] = u;
    }
#if HAVE_TDM
    if (t < 32) tdm_wait();
#else
#pragma unroll
    for (int i = 0; i < 16; ++i) {
        int q = t + i * 128;
        int row = q >> 5, c = q & 31;
        int gr = n0 + row; if (gr >= N) gr = N - 1;
        ((uint4*)Hs)[q] = ((const uint4*)(hb + (size_t)gr * SS))[c];
    }
#endif
    __syncthreads();
    int wave = t >> 5, lane = t & 31;
    int rowbase = wave * 16;
    int rh = lane >> 4, col16 = lane & 15;
    bool full = (n0 + 64 <= N);
    for (int ct = 0; ct < 16; ++ct) {
        v8f air = {}, aiz = {}, ain = {}, ahr = {}, ahz = {}, ahn = {};
#pragma unroll
        for (int kk = 0; kk < 8; ++kk) {
            int kb = kk * 32;
            v16bf ag = load_frag_a(Ags, rowbase, SS, kb, lane);
            v16bf ah = load_frag_a(Hs, rowbase, SS, kb, lane);
            air = wmma_bf16(ag, load_frag_b(wihb, ct * 16,           SS, kb, lane), air);
            aiz = wmma_bf16(ag, load_frag_b(wihb, SS + ct * 16,      SS, kb, lane), aiz);
            ain = wmma_bf16(ag, load_frag_b(wihb, 2 * SS + ct * 16,  SS, kb, lane), ain);
            ahr = wmma_bf16(ah, load_frag_b(whhb, ct * 16,           SS, kb, lane), ahr);
            ahz = wmma_bf16(ah, load_frag_b(whhb, SS + ct * 16,      SS, kb, lane), ahz);
            ahn = wmma_bf16(ah, load_frag_b(whhb, 2 * SS + ct * 16,  SS, kb, lane), ahn);
        }
        int col = ct * 16 + col16;
        float bir = bih[col], biz = bih[SS + col], bin = bih[2 * SS + col];
        float bhr = bhh[col], bhz = bhh[SS + col], bhn = bhh[2 * SS + col];
        size_t base = (size_t)(n0 + rowbase + 8 * rh) * SS + col;
        if (full) {
#pragma unroll
            for (int i = 0; i < 8; ++i) {
                float r  = sigmoidf_(air[i] + bir + ahr[i] + bhr);
                float z  = sigmoidf_(aiz[i] + biz + ahz[i] + bhz);
                float ng = tanhf(ain[i] + bin + r * (ahn[i] + bhn));
                float ho = hf[base + (size_t)i * SS];
                float hn = (1.0f - z) * ng + z * ho;
                hf_out[base + (size_t)i * SS] = hn;
                hb_out[base + (size_t)i * SS] = (bf16_t)hn;
            }
        } else {
#pragma unroll
            for (int i = 0; i < 8; ++i) {
                int node = n0 + rowbase + i + 8 * rh;
                if (node < N) {
                    float r  = sigmoidf_(air[i] + bir + ahr[i] + bhr);
                    float z  = sigmoidf_(aiz[i] + biz + ahz[i] + bhz);
                    float ng = tanhf(ain[i] + bin + r * (ahn[i] + bhn));
                    float ho = hf[(size_t)node * SS + col];
                    float hn = (1.0f - z) * ng + z * ho;
                    hf_out[(size_t)node * SS + col] = hn;
                    hb_out[(size_t)node * SS + col] = (bf16_t)hn;
                }
            }
        }
    }
}

// ---------------- fused per-node MLP: relu(relu([h,x]@l0^T+b0)@l1^T+b1) ----------
__global__ void __launch_bounds__(128) mlp_kernel(const bf16_t* __restrict__ hb,
                                                  const bf16_t* __restrict__ xb,
                                                  const bf16_t* __restrict__ l0wb,
                                                  const float* __restrict__ l0b,
                                                  const bf16_t* __restrict__ l1wb,
                                                  const float* __restrict__ l1b,
                                                  float* __restrict__ out, int N) {
    __shared__ bf16_t Hpart[64 * SS];
    __shared__ bf16_t Xpart[64 * ANN_];
    __shared__ bf16_t Y1[64 * MLPH];
    int t = threadIdx.x;
    int n0 = blockIdx.x * 64;
#if HAVE_TDM
    if (t < 32) {
        tdm_load_2d_bf16(hb + (size_t)n0 * SS, (unsigned)(uintptr_t)Hpart,
                         (unsigned)(N - n0), 64u, (unsigned)SS);
        tdm_load_2d_bf16(xb + (size_t)n0 * ANN_, (unsigned)(uintptr_t)Xpart,
                         (unsigned)(N - n0), 64u, (unsigned)ANN_);
        tdm_wait();
    }
    __syncthreads();
#else
#pragma unroll
    for (int i = 0; i < 16; ++i) {
        int q = t + i * 128;
        int row = q >> 5, c = q & 31;
        int gr = n0 + row; if (gr >= N) gr = N - 1;
        ((uint4*)Hpart)[q] = ((const uint4*)(hb + (size_t)gr * SS))[c];
    }
#pragma unroll
    for (int i = 0; i < 8; ++i) {
        int q = t + i * 128;
        int row = q >> 4, c = q & 15;
        int gr = n0 + row; if (gr >= N) gr = N - 1;
        ((uint4*)Xpart)[q] = ((const uint4*)(xb + (size_t)gr * ANN_))[c];
    }
    __syncthreads();
#endif
    int wave = t >> 5, lane = t & 31;
    int rowbase = wave * 16;
    int rh = lane >> 4, col16 = lane & 15;
    bool full = (n0 + 64 <= N);
    // stage 1: y1 = relu(cat @ l0^T + b0)   [64 x 256]; B pipelined from global
    for (int ct = 0; ct < 16; ++ct) {
        v8f acc = {};
        v16bf b = load_frag_b(l0wb, ct * 16, CATW, 0, lane);
#pragma unroll
        for (int kk = 0; kk < 12; ++kk) {
            v16bf bn = (kk < 11) ? load_frag_b(l0wb, ct * 16, CATW, (kk + 1) * 32, lane) : b;
            v16bf a = (kk < 8) ? load_frag_a(Hpart, rowbase, SS, kk * 32, lane)
                               : load_frag_a(Xpart, rowbase, ANN_, kk * 32 - SS, lane);
            acc = wmma_bf16(a, b, acc);
            b = bn;
        }
        int col = ct * 16 + col16;
        float bias = l0b[col];
#pragma unroll
        for (int i = 0; i < 8; ++i) {
            int lrow = rowbase + i + 8 * rh;
            Y1[lrow * MLPH + col] = (bf16_t)fmaxf(acc[i] + bias, 0.0f);
        }
    }
    __syncthreads();
    // stage 2: out = relu(y1 @ l1^T + b1)   [64 x 128]
    for (int ct = 0; ct < 8; ++ct) {
        v8f acc = {};
        v16bf b = load_frag_b(l1wb, ct * 16, MLPH, 0, lane);
#pragma unroll
        for (int kk = 0; kk < 8; ++kk) {
            v16bf bn = (kk < 7) ? load_frag_b(l1wb, ct * 16, MLPH, (kk + 1) * 32, lane) : b;
            v16bf a = load_frag_a(Y1, rowbase, MLPH, kk * 32, lane);
            acc = wmma_bf16(a, b, acc);
            b = bn;
        }
        int col = ct * 16 + col16;
        float bias = l1b[col];
        float* op = out + (size_t)(n0 + rowbase + 8 * rh) * OUTW + col;
        if (full) {
#pragma unroll
            for (int i = 0; i < 8; ++i) op[(size_t)i * OUTW] = fmaxf(acc[i] + bias, 0.0f);
        } else {
#pragma unroll
            for (int i = 0; i < 8; ++i) {
                int node = n0 + rowbase + i + 8 * rh;
                if (node < N) out[(size_t)node * OUTW + col] = fmaxf(acc[i] + bias, 0.0f);
            }
        }
    }
}

extern "C" void kernel_launch(void* const* d_in, const int* in_sizes, int n_in,
                              void* d_out, int out_size, void* d_ws, size_t ws_size,
                              hipStream_t stream) {
    const float* x    = (const float*)d_in[0];
    const int*   ei   = (const int*)d_in[1];
    const float* W    = (const float*)d_in[2];
    const float* wih  = (const float*)d_in[3];
    const float* whh  = (const float*)d_in[4];
    const float* bih  = (const float*)d_in[5];
    const float* bhh  = (const float*)d_in[6];
    const float* l0w  = (const float*)d_in[7];
    const float* l0b  = (const float*)d_in[8];
    const float* l1w  = (const float*)d_in[9];
    const float* l1b  = (const float*)d_in[10];
    float* out = (float*)d_out;

    const int N = in_sizes[0] / ANN_;       // 100000
    const int E = in_sizes[1] / 2;          // 1600000
    const int L = 5;

    // workspace carve-out (256B aligned)
    char* ws = (char*)d_ws;
    size_t cur = 0;
    auto carve = [&](size_t bytes) -> void* {
        void* p = ws + cur;
        cur = (cur + bytes + 255) & ~(size_t)255;
        return p;
    };
    bf16_t* WT    = (bf16_t*)carve((size_t)L * SS * SS * 2);
    bf16_t* wihb  = (bf16_t*)carve((size_t)3 * SS * SS * 2);
    bf16_t* whhb  = (bf16_t*)carve((size_t)3 * SS * SS * 2);
    bf16_t* l0wb  = (bf16_t*)carve((size_t)MLPH * CATW * 2);
    bf16_t* l1wb  = (bf16_t*)carve((size_t)OUTW * MLPH * 2);
    float*  hf0   = (float*)carve((size_t)N * SS * 4);
    float*  hf1   = (float*)carve((size_t)N * SS * 4);
    bf16_t* hb0   = (bf16_t*)carve((size_t)N * SS * 2);
    bf16_t* hb1   = (bf16_t*)carve((size_t)N * SS * 2);
    bf16_t* xb    = (bf16_t*)carve((size_t)N * ANN_ * 2);
    bf16_t* mb    = (bf16_t*)carve((size_t)N * SS * 2);
    float*  agg   = (float*)carve((size_t)N * SS * 4);
    float*  hf[2]  = {hf0, hf1};
    bf16_t* hbp[2] = {hb0, hb1};

    const int NW = 5 * SS * SS + 6 * SS * SS + MLPH * CATW + OUTW * MLPH;
    convert_weights_kernel<<<(NW + 255) / 256, 256, 0, stream>>>(W, wih, whh, l0w, l1w,
                                                                 WT, wihb, whhb, l0wb, l1wb);
    init_h_kernel<<<((size_t)N * SS + 255) / 256, 256, 0, stream>>>(x, hf0, hb0, xb, N);

    const int NB = (N + 63) / 64;
    int fin = 0;
    for (int l = 0; l < L; ++l) {
        int curb = l & 1, nxt = curb ^ 1;
        gemm_m_kernel<<<NB, 128, 0, stream>>>(hbp[curb], WT + (size_t)l * SS * SS, mb, N);
        (void)hipMemsetAsync(agg, 0, (size_t)N * SS * 4, stream);
        scatter_kernel<<<((size_t)E * 32 + 255) / 256, 256, 0, stream>>>(ei, mb, agg, E);
        gru_kernel<<<NB, 128, 0, stream>>>(agg, hbp[curb], hf[curb], wihb, whhb, bih, bhh,
                                           hf[nxt], hbp[nxt], N);
        fin = nxt;
    }
    mlp_kernel<<<NB, 128, 0, stream>>>(hbp[fin], xb, l0wb, l0b, l1wb, l1b, out, N);
}
